// MTAAttention_79121887527347
// MI455X (gfx1250) — matrix operand order repaired and checked by reading, not compiled, not run on hardware
//
#include <hip/hip_runtime.h>
#include <math.h>

#define H 16
#define S 2048
#define D 64

typedef __attribute__((ext_vector_type(16))) __bf16 v16bf;
typedef __attribute__((ext_vector_type(8)))  float v8f;

union BF16x16 { unsigned u[8]; v16bf v; };

__device__ __forceinline__ unsigned short f2bf(float x) {
  unsigned u = __float_as_uint(x);
  u += 0x7FFFu + ((u >> 16) & 1u);          // round-to-nearest-even
  return (unsigned short)(u >> 16);
}
__device__ __forceinline__ float bf2f(unsigned short h) {
  return __uint_as_float(((unsigned)h) << 16);
}
__device__ __forceinline__ unsigned packbf(float a, float b) {
  return (unsigned)f2bf(a) | ((unsigned)f2bf(b) << 16);
}

// ---------------------------------------------------------------------------
// Kernel 0: fp32 -> bf16 repack into WMMA-friendly layouts.
// qpk[h][s][dp]   : Q d-pairs  (A fragment of QK^T, contraction = d)
// ktp[h][dp][s]   : K^T d-pairs (B fragment of QK^T; lanes walk seq -> coalesced)
// vpk[h][s/2][d]  : V seq-pairs (B fragment of PV;   lanes walk d   -> coalesced)
// ---------------------------------------------------------------------------
__global__ __launch_bounds__(256) void k_convert(const float* __restrict__ xq,
                                                 const float* __restrict__ xk,
                                                 const float* __restrict__ xv,
                                                 unsigned* __restrict__ qpk,
                                                 unsigned* __restrict__ ktp,
                                                 unsigned* __restrict__ vpk) {
  int idx = blockIdx.x * blockDim.x + threadIdx.x;
  const int total = H * S * (D / 2);       // == H*32*S == H*(S/2)*D == 1M
  if (idx >= total) return;

  { // qpk
    int dp = idx & 31; int hs = idx >> 5; int s = hs & (S - 1); int h = hs >> 11;
    const float* q = xq + ((size_t)h * S + s) * D + 2 * dp;
    qpk[idx] = packbf(q[0], q[1]);
  }
  { // ktp
    int s = idx & (S - 1); int hd = idx >> 11; int dp = hd & 31; int h = hd >> 5;
    const float* k = xk + ((size_t)h * S + s) * D + 2 * dp;
    ktp[idx] = packbf(k[0], k[1]);
  }
  { // vpk
    int d = idx & (D - 1); int hs = idx >> 6; int sp = hs & (S / 2 - 1); int h = hs >> 10;
    const float* v = xv + ((size_t)h * S + 2 * sp) * D + d;
    vpk[idx] = packbf(v[0], v[D]);
  }
}

// ---------------------------------------------------------------------------
// Kernel 1: scores = (Q K^T) * rsqrt(64), masked (k>q -> 0), bf16 out.
// One wave per 16x16 tile; 2x v_wmma_f32_16x16x32_bf16 over D=64.
// ---------------------------------------------------------------------------
__global__ __launch_bounds__(128) void k_qk(const unsigned* __restrict__ qpk,
                                            const unsigned* __restrict__ ktp,
                                            unsigned short* __restrict__ s0) {
  int lane = threadIdx.x & 31;
  int wave = threadIdx.x >> 5;
  int t = blockIdx.x * 4 + wave;           // tile id over H*128*128
  int h  = t >> 14;
  int rem = t & 16383;
  int q0 = (rem >> 7) * 16, k0 = (rem & 127) * 16;
  int lo = lane & 15, hi = lane >> 4;

  if (k0 > q0 + 15) {                      // fully above diagonal -> zero fill
#pragma unroll
    for (int j = 0; j < 8; ++j) {
      int q = q0 + j + 8 * hi, k = k0 + lo;
      s0[((size_t)h * S + q) * S + k] = 0;
    }
    return;
  }

  BF16x16 a0, a1, b0, b1;
  const unsigned* qrow = qpk + ((size_t)h * S + q0 + lo) * 32;
#pragma unroll
  for (int i = 0; i < 8; ++i) {            // A 16x32 bf16 layout (d-pairs)
    int dp = i + (hi << 3);
    a0.u[i] = qrow[dp];
    a1.u[i] = qrow[dp + 16];
  }
#pragma unroll
  for (int i = 0; i < 8; ++i) {            // B 32x16 bf16 layout
    int pidx = i + (hi << 3);
    b0.u[i] = ktp[((size_t)h * 32 + pidx) * S + k0 + lo];
    b1.u[i] = ktp[((size_t)h * 32 + pidx + 16) * S + k0 + lo];
  }

  v8f acc = {};
  acc = __builtin_amdgcn_wmma_f32_16x16x32_bf16(false, a0.v, false, b0.v,
                                                (short)0, acc, false, false);
  acc = __builtin_amdgcn_wmma_f32_16x16x32_bf16(false, a1.v, false, b1.v,
                                                (short)0, acc, false, false);
#pragma unroll
  for (int j = 0; j < 8; ++j) {
    int q = q0 + j + 8 * hi, k = k0 + lo;
    float v = (k <= q) ? acc[j] * 0.125f : 0.f;
    s0[((size_t)h * S + q) * S + k] = f2bf(v);
  }
}

// ---------------------------------------------------------------------------
// Kernel 2: 6x11 depthwise conv (pad q:(5,0), k:(5,5)) + pre-softmax head mix.
// Block = 16 waves (wave = head), tile = 16q x 32k.  Fusing the mix avoids a
// full 256MB round trip of the un-mixed conv output.
// ---------------------------------------------------------------------------
__global__ __launch_bounds__(512) void k_conv_mix(const unsigned short* __restrict__ s0,
                                                  const float* __restrict__ ker,
                                                  const float* __restrict__ wpsm,
                                                  float* __restrict__ s1) {
  __shared__ unsigned short in_t[16][21][42];   // 28.2 KB  (q0-5..q0+15, k0-5..k0+36)
  __shared__ unsigned short c_t[16][16][32];    // 16.4 KB  conv output per head
  __shared__ float wk[16][66];
  __shared__ float wmix[256];

  int q0 = blockIdx.y * 16, k0 = blockIdx.x * 32;
  if (k0 > q0 + 15) return;                     // tile entirely above diagonal
  int tid = threadIdx.x;
  int lane = tid & 31, h = tid >> 5;

  for (int i = tid; i < 16 * 66; i += 512) wk[i / 66][i % 66] = ker[i];
  if (tid < 256) wmix[tid] = wpsm[tid];

  for (int idx = lane; idx < 21 * 42; idx += 32) {
    int r = idx / 42, c = idx % 42;
    int q = q0 - 5 + r, k = k0 - 5 + c;
    unsigned short v = 0;
    if (q >= 0 && k >= 0 && k < S) v = s0[((size_t)h * S + q) * S + k];
    in_t[h][r][c] = v;
  }
  __syncthreads();

  // conv: lane owns one output column, 16 rows; column-strip register reuse
  float o[16];
#pragma unroll
  for (int m = 0; m < 16; ++m) o[m] = 0.f;
#pragma unroll
  for (int b = 0; b < 11; ++b) {
    float col[21];
#pragma unroll
    for (int r = 0; r < 21; ++r) col[r] = bf2f(in_t[h][r][lane + b]);
#pragma unroll
    for (int a = 0; a < 6; ++a) {
      float w = wk[h][a * 11 + b];
#pragma unroll
      for (int m = 0; m < 16; ++m) o[m] = fmaf(w, col[m + a], o[m]);
    }
  }
#pragma unroll
  for (int m = 0; m < 16; ++m) c_t[h][m][lane] = f2bf(o[m]);
  __syncthreads();

  // head mix: wave g, column lane
  int g = h;
#pragma unroll 4
  for (int m = 0; m < 16; ++m) {
    float acc = 0.f;
#pragma unroll
    for (int hh = 0; hh < 16; ++hh)
      acc = fmaf(wmix[g * 16 + hh], bf2f(c_t[hh][m][lane]), acc);
    s1[((size_t)g * S + q0 + m) * S + k0 + lane] = acc;
  }
}

// ---------------------------------------------------------------------------
// Kernel 3: per-row softmax stats (max, 1/sum-exp) over k<=q. Wave32 reduce.
// ---------------------------------------------------------------------------
__global__ __launch_bounds__(256) void k_stats(const float* __restrict__ s1,
                                               float* __restrict__ mrow,
                                               float* __restrict__ linv) {
  int lane = threadIdx.x & 31;
  int wave = threadIdx.x >> 5;
  int row = blockIdx.x * 8 + wave;              // g*S + q
  int q = row & (S - 1);
  const float* r = s1 + (size_t)row * S;

  float m = -1e30f, l = 0.f;
  for (int k = lane; k <= q; k += 32) {
    float x = r[k];
    if (x > m) { l = l * __expf(m - x) + 1.f; m = x; }
    else       { l += __expf(x - m); }
  }
#pragma unroll
  for (int off = 16; off; off >>= 1) {
    float m2 = __shfl_xor(m, off, 32);
    float l2 = __shfl_xor(l, off, 32);
    float mn = fmaxf(m, m2);
    l = l * __expf(m - mn) + l2 * __expf(m2 - mn);
    m = mn;
  }
  if (lane == 0) { mrow[row] = m; linv[row] = 1.f / l; }
}

// ---------------------------------------------------------------------------
// Kernel 4: softmax -> 3x3 conv -> re-mask -> post head mix -> PV (WMMA bf16).
// Block = 16 waves (wave = output head g), one 16-query tile, streamed over k.
// ---------------------------------------------------------------------------
__global__ __launch_bounds__(512) void k_epilogue(const float* __restrict__ s1,
                                                  const float* __restrict__ mrow,
                                                  const float* __restrict__ linv,
                                                  const float* __restrict__ ker2,
                                                  const float* __restrict__ wposm,
                                                  const unsigned* __restrict__ vpk,
                                                  float* __restrict__ out) {
  __shared__ unsigned short p_t[16][18][36];    // 20.7 KB softmax probs (+halos)
  __shared__ float c_t[16][16][32];             // 32.8 KB conv3x3 per head
  __shared__ float wk2[16][9];
  __shared__ float wmix2[256];
  __shared__ float st_m[16][18];
  __shared__ float st_i[16][18];

  int q0 = blockIdx.x * 16;
  int tid = threadIdx.x;
  int lane = tid & 31, h = tid >> 5;
  int lo = lane & 15, hi = lane >> 4;

  for (int i = tid; i < 144; i += 512) wk2[i / 9][i % 9] = ker2[i];
  if (tid < 256) wmix2[tid] = wposm[tid];
  if (lane < 18) {
    int q = q0 - 2 + lane;
    float m = 0.f, li = 0.f;
    if (q >= 0) { m = mrow[h * S + q]; li = linv[h * S + q]; }
    st_m[h][lane] = m; st_i[h][lane] = li;
  }
  __syncthreads();

  v8f acc0 = {}, acc1 = {}, acc2 = {}, acc3 = {};
  const unsigned* vbase = vpk + (size_t)h * (S / 2) * D;

  for (int k0 = 0; k0 <= q0 + 15; k0 += 32) {
    // softmax probabilities with halos (rows q0-2..q0+15, cols k0-1..k0+32)
    for (int idx = lane; idx < 18 * 34; idx += 32) {
      int rr = idx / 34, cc = idx % 34;
      int q = q0 - 2 + rr, k = k0 - 1 + cc;
      float p = 0.f;
      if (q >= 0 && k >= 0 && k <= q && k < S) {
        float x = s1[((size_t)h * S + q) * S + k];
        p = __expf(x - st_m[h][rr]) * st_i[h][rr];
      }
      p_t[h][rr][cc] = f2bf(p);
    }

    // 3x3 conv (pad q:(2,0), k:(1,1)), then causal re-zero
    float o[16];
#pragma unroll
    for (int m = 0; m < 16; ++m) o[m] = 0.f;
#pragma unroll
    for (int b = 0; b < 3; ++b) {
      float col[18];
#pragma unroll
      for (int r = 0; r < 18; ++r) col[r] = bf2f(p_t[h][r][lane + b]);
#pragma unroll
      for (int a = 0; a < 3; ++a) {
        float w = wk2[h][a * 3 + b];
#pragma unroll
        for (int m = 0; m < 16; ++m) o[m] = fmaf(w, col[m + a], o[m]);
      }
    }
#pragma unroll
    for (int m = 0; m < 16; ++m)
      c_t[h][m][lane] = ((k0 + lane) <= (q0 + m)) ? o[m] : 0.f;
    __syncthreads();

    // head mix while assembling the A fragment (16x32 bf16, k-pairs)
    int g = h;
    BF16x16 a;
#pragma unroll
    for (int i = 0; i < 8; ++i) {
      int kp = i + (hi << 3);
      int k = 2 * kp;
      float f0 = 0.f, f1 = 0.f;
#pragma unroll
      for (int hh = 0; hh < 16; ++hh) {
        float w = wmix2[g * 16 + hh];
        f0 = fmaf(w, c_t[hh][lo][k], f0);
        f1 = fmaf(w, c_t[hh][lo][k + 1], f1);
      }
      a.u[i] = packbf(f0, f1);
    }

    // PV: 4 WMMAs cover d = 0..63
    BF16x16 b;
#pragma unroll
    for (int i = 0; i < 8; ++i) {
      int kk2 = (k0 >> 1) + i + (hi << 3);
      b.u[i] = vbase[(size_t)kk2 * D + lo];
    }
    acc0 = __builtin_amdgcn_wmma_f32_16x16x32_bf16(false, a.v, false, b.v, (short)0, acc0, false, false);
#pragma unroll
    for (int i = 0; i < 8; ++i) {
      int kk2 = (k0 >> 1) + i + (hi << 3);
      b.u[i] = vbase[(size_t)kk2 * D + 16 + lo];
    }
    acc1 = __builtin_amdgcn_wmma_f32_16x16x32_bf16(false, a.v, false, b.v, (short)0, acc1, false, false);
#pragma unroll
    for (int i = 0; i < 8; ++i) {
      int kk2 = (k0 >> 1) + i + (hi << 3);
      b.u[i] = vbase[(size_t)kk2 * D + 32 + lo];
    }
    acc2 = __builtin_amdgcn_wmma_f32_16x16x32_bf16(false, a.v, false, b.v, (short)0, acc2, false, false);
#pragma unroll
    for (int i = 0; i < 8; ++i) {
      int kk2 = (k0 >> 1) + i + (hi << 3);
      b.u[i] = vbase[(size_t)kk2 * D + 48 + lo];
    }
    acc3 = __builtin_amdgcn_wmma_f32_16x16x32_bf16(false, a.v, false, b.v, (short)0, acc3, false, false);
    __syncthreads();                       // protect c_t for next iteration
  }

  // out layout [B,S,H,D] = [q][g][d]
#pragma unroll
  for (int j = 0; j < 8; ++j) {
    int q = q0 + j + 8 * hi;
    float* orow = out + ((size_t)q * H + h) * D;
    orow[lo]      = acc0[j];
    orow[16 + lo] = acc1[j];
    orow[32 + lo] = acc2[j];
    orow[48 + lo] = acc3[j];
  }
}

// ---------------------------------------------------------------------------
extern "C" void kernel_launch(void* const* d_in, const int* in_sizes, int n_in,
                              void* d_out, int out_size, void* d_ws, size_t ws_size,
                              hipStream_t stream) {
  (void)in_sizes; (void)n_in; (void)out_size; (void)ws_size;
  const float* xq    = (const float*)d_in[0];
  const float* xk    = (const float*)d_in[1];
  const float* xv    = (const float*)d_in[2];
  // d_in[3] = mask: pure causal triu(-inf); handled analytically.
  const float* ker1  = (const float*)d_in[4];   // [16,1,6,11]
  const float* ker2  = (const float*)d_in[5];   // [16,1,3,3]
  const float* wpsm  = (const float*)d_in[6];   // [16,16]
  const float* wposm = (const float*)d_in[7];   // [16,16]

  char* ws = (char*)d_ws;
  size_t off = 0;
  auto alloc = [&](size_t bytes) -> void* {
    void* p = ws + off;
    off += (bytes + 255) & ~(size_t)255;
    return p;
  };
  unsigned*       qpk  = (unsigned*)alloc((size_t)H * S * 32 * 4);        //  4 MB
  unsigned*       ktp  = (unsigned*)alloc((size_t)H * 32 * S * 4);        //  4 MB
  unsigned*       vpk  = (unsigned*)alloc((size_t)H * (S / 2) * D * 4);   //  4 MB
  unsigned short* s0   = (unsigned short*)alloc((size_t)H * S * S * 2);   // 128 MB
  float*          s1   = (float*)alloc((size_t)H * S * S * 4);            // 256 MB
  float*          mrw  = (float*)alloc((size_t)H * S * 4);
  float*          lnv  = (float*)alloc((size_t)H * S * 4);
  float*          outp = (float*)d_out;

  k_convert<<<(H * S * 32 + 255) / 256, 256, 0, stream>>>(xq, xk, xv, qpk, ktp, vpk);
  k_qk<<<(H * 128 * 128) / 4, 128, 0, stream>>>(qpk, ktp, s0);
  k_conv_mix<<<dim3(64, 128), 512, 0, stream>>>(s0, ker1, wpsm, s1);
  k_stats<<<(H * S) / 8, 256, 0, stream>>>(s1, mrw, lnv);
  k_epilogue<<<128, 512, 0, stream>>>(s1, mrw, lnv, ker2, wposm, vpk, outp);
}
